// VectorQuantizer_26130581029055
// MI455X (gfx1250) — matrix operand order repaired and checked by reading, not compiled, not run on hardware
//
#include <hip/hip_runtime.h>
#include <math.h>

// ---------------- problem constants ----------------
#define N_DIM 32768
#define K_DIM 8192
#define D_DIM 768
#define DECAY 0.99f
#define ONE_M_DECAY (1.0f - 0.99f)
#define BETA 0.25f

// ---------------- output layout (floats) ----------------
#define OFF_Q      0                   // N*D
#define OFF_IDX    25165824            // N
#define OFF_LOSS   25198592
#define OFF_COMMIT 25198593
#define OFF_CODE   25198594
#define OFF_PERP   25198595
#define OFF_USAGE  25198596
#define OFF_NW     25198597            // K*D
#define OFF_NCS    31490053            // K
#define OFF_NEW    31498245            // K*D

// ---------------- workspace layout (bytes) ----------------
#define WS_WB    0           // K*D bf16 (ushort)            = 12582912
#define WS_WN    12582912    // K floats                     = 32768
#define WS_SUMS  12615680    // K*D floats (segment sums)    = 25165824
#define WS_CNT   37781504    // K floats                     = 32768
#define WS_ACC   37814272    // accumulators (floats)        = 256
#define WS_IDX   37814528    // N ints                       = 131072

typedef __attribute__((ext_vector_type(16))) __bf16 v16bf;
typedef __attribute__((ext_vector_type(8)))  __bf16 v8bf;
typedef __attribute__((ext_vector_type(8)))  float  v8f;
typedef __attribute__((ext_vector_type(4)))  unsigned v4u;
typedef __attribute__((ext_vector_type(8)))  int    v8i;
typedef __attribute__((ext_vector_type(4)))  int    v4i;

#if __has_builtin(__builtin_amdgcn_tensor_load_to_lds) && \
    __has_builtin(__builtin_amdgcn_s_wait_tensorcnt)
#define USE_TDM 1
#endif

__device__ __forceinline__ unsigned short f2bf(float f) {
    union { float f; unsigned u; } c; c.f = f;
    unsigned u = c.u;
    u += 0x7fffu + ((u >> 16) & 1u);       // round to nearest even
    return (unsigned short)(u >> 16);
}

__device__ __forceinline__ unsigned pcg(unsigned v) {
    unsigned state = v * 747796405u + 2891336453u;
    unsigned word = ((state >> ((state >> 28) + 4u)) ^ state) * 277803737u;
    return (word >> 22) ^ word;
}

// =====================================================================
// Kernel 1: weight f32 -> bf16, plus ||w||^2 per codeword
// =====================================================================
__global__ void vq_prep_w(const float* __restrict__ W,
                          unsigned short* __restrict__ Wb,
                          float* __restrict__ wnorm) {
    __shared__ float red[256];
    const int k = blockIdx.x, tid = threadIdx.x;
    float ss = 0.f;
    for (int d = tid; d < D_DIM; d += 256) {
        float w = W[(size_t)k * D_DIM + d];
        Wb[(size_t)k * D_DIM + d] = f2bf(w);
        ss += w * w;
    }
    red[tid] = ss; __syncthreads();
    for (int s = 128; s > 0; s >>= 1) {
        if (tid < s) red[tid] += red[tid + s];
        __syncthreads();
    }
    if (tid == 0) wnorm[k] = red[0];
}

// =====================================================================
// B-tile staging: Tensor Data Mover (async DMA, one issue per block from
// wave 0), falling back to a cooperative vector-copy loop if the TDM
// builtin is unavailable.  Tile = 32 codewords x 768 bf16 = 48 KB.
// 1-D D#: data_size=8B, tensor_dim0 = tile_dim0 = 6144, type=2 (image).
// This toolchain exposes the 6-arg TDM builtin:
//   (uint32x4 g0, int32x8 g1, int32x4 g2, int32x4 g3, int32x8, i32 cpol)
// =====================================================================
#define BTILE_USHORTS (32 * D_DIM)          // 24576
#define BTILE_BYTES   (BTILE_USHORTS * 2)   // 49152

__device__ __forceinline__ void stage_b(unsigned short* __restrict__ dstLds,
                                        const unsigned short* __restrict__ srcGlobal,
                                        int tid) {
#ifdef USE_TDM
    if (tid < 32) {
        unsigned lds_addr = (unsigned)(unsigned long long)(uintptr_t)dstLds;
        unsigned long long ga = (unsigned long long)(uintptr_t)srcGlobal;
        v4u g0;
        g0.x = 1u;                                           // count=1, user desc
        g0.y = lds_addr;                                     // lds_addr
        g0.z = (unsigned)(ga & 0xffffffffu);                 // global_addr[31:0]
        g0.w = (unsigned)((ga >> 32) & 0x01ffffffu) | (2u << 30); // addr[56:32] | type=2
        v8i g1;
        g1[0] = (int)(3u << 16);          // wg_mask=0, data_size=3 (8B)
        g1[1] = (int)(6144u << 16);       // tensor_dim0[15:0] @ bits 63:48
        g1[2] = 0;                        // tensor_dim0[31:16]=0, tensor_dim1=0
        g1[3] = (int)(6144u << 16);       // tile_dim0 @ bits 127:112
        g1[4] = 0; g1[5] = 0; g1[6] = 0; g1[7] = 0;
        v4i z4 = {0, 0, 0, 0};
        v8i z8 = {0, 0, 0, 0, 0, 0, 0, 0};
        __builtin_amdgcn_tensor_load_to_lds(g0, g1, z4, z4, z8, 0);
    }
#else
    const uint4* s = (const uint4*)srcGlobal;
    uint4* d = (uint4*)dstLds;
    for (int i = tid; i < BTILE_BYTES / 16; i += 128) d[i] = s[i];
#endif
}

__device__ __forceinline__ void stage_b_wait(int tid) {
#ifdef USE_TDM
    if (tid < 32) __builtin_amdgcn_s_wait_tensorcnt((short)0);
#endif
}

// =====================================================================
// Kernel 2 (main): WMMA bf16 nearest-codeword search, 2M x 2N blocked.
// Block = 128 threads = 4 waves; each wave owns 32 rows (2 M-tiles);
// block covers 128 rows.  Per iteration 32 codewords are staged (TDM),
// each wave runs 4 independent WMMA accumulator chains over full D=768
// (24 chunks): per chunk 4 A-loads + 4 B-loads -> 4 WMMAs (2 ds/WMMA).
// LDS: A = 128x768 bf16 (192KB), B = 2 x 32x768 bf16 (96KB) = 288KB.
// =====================================================================
__global__ __launch_bounds__(128) void vq_argmin_kernel(
        const float* __restrict__ X,
        const unsigned short* __restrict__ Wb,
        const float* __restrict__ wnorm,
        int* __restrict__ idx_i,
        float* __restrict__ idx_f) {
    extern __shared__ char smem[];
    unsigned short* As = (unsigned short*)smem;          // [128][768]
    unsigned short* Bs = As + 128 * D_DIM;               // [2][32][768]
    const int tid = threadIdx.x;

    // issue async load of B tile 0 (overlaps with A staging)
    stage_b(Bs, Wb, tid);

    // ---- stage A: 128 rows of X, f32 -> bf16 ----
    {
        const float4* xs = (const float4*)(X + (size_t)blockIdx.x * 128 * D_DIM);
        for (int i = tid; i < (128 * D_DIM) / 4; i += 128) {
            float4 v = xs[i];
            uint2 p;
            p.x = (unsigned)f2bf(v.x) | ((unsigned)f2bf(v.y) << 16);
            p.y = (unsigned)f2bf(v.z) | ((unsigned)f2bf(v.w) << 16);
            *(uint2*)(As + (size_t)i * 4) = p;
        }
    }
    stage_b_wait(tid);
    __syncthreads();

    const int wave = tid >> 5, lane = tid & 31;
    const int lh = lane >> 4, l15 = lane & 15;
    const unsigned short* Arow0 = As + (size_t)(wave * 32 + l15) * D_DIM;
    const unsigned short* Arow1 = Arow0 + (size_t)16 * D_DIM;

    float best[2][8]; int bestn[2][8];
#pragma unroll
    for (int mt = 0; mt < 2; ++mt)
#pragma unroll
        for (int r = 0; r < 8; ++r) { best[mt][r] = 3.4e38f; bestn[mt][r] = 0; }

    const int NT = K_DIM / 32;   // 256 iterations of 32 codewords
    for (int t = 0; t < NT; ++t) {
        const int cur = t & 1;
        if (t + 1 < NT)
            stage_b(Bs + (size_t)(cur ^ 1) * BTILE_USHORTS,
                    Wb + (size_t)(t + 1) * BTILE_USHORTS, tid);

        const unsigned short* Bbuf = Bs + (size_t)cur * BTILE_USHORTS;
        const unsigned short* Brow0 = Bbuf + (size_t)l15 * D_DIM + lh * 16;
        const unsigned short* Brow1 = Bbuf + (size_t)(16 + l15) * D_DIM + lh * 16;

        v8f acc00 = {}, acc01 = {}, acc10 = {}, acc11 = {};
#pragma unroll
        for (int dc = 0; dc < D_DIM; dc += 32) {
            const unsigned short* pa0 = Arow0 + dc + lh * 16;
            const unsigned short* pa1 = Arow1 + dc + lh * 16;
            v8bf a0lo = *(const v8bf*)pa0;
            v8bf a0hi = *(const v8bf*)(pa0 + 8);
            v8bf a1lo = *(const v8bf*)pa1;
            v8bf a1hi = *(const v8bf*)(pa1 + 8);
            v8bf b0lo = *(const v8bf*)(Brow0 + dc);
            v8bf b0hi = *(const v8bf*)(Brow0 + dc + 8);
            v8bf b1lo = *(const v8bf*)(Brow1 + dc);
            v8bf b1hi = *(const v8bf*)(Brow1 + dc + 8);
            v16bf a0, a1, b0, b1;
#pragma unroll
            for (int i = 0; i < 8; ++i) {
                a0[i] = a0lo[i]; a0[i + 8] = a0hi[i];
                a1[i] = a1lo[i]; a1[i + 8] = a1hi[i];
                b0[i] = b0lo[i]; b0[i + 8] = b0hi[i];
                b1[i] = b1lo[i]; b1[i + 8] = b1hi[i];
            }
            acc00 = __builtin_amdgcn_wmma_f32_16x16x32_bf16(
                false, a0, false, b0, (short)0, acc00, false, false);
            acc01 = __builtin_amdgcn_wmma_f32_16x16x32_bf16(
                false, a0, false, b1, (short)0, acc01, false, false);
            acc10 = __builtin_amdgcn_wmma_f32_16x16x32_bf16(
                false, a1, false, b0, (short)0, acc10, false, false);
            acc11 = __builtin_amdgcn_wmma_f32_16x16x32_bf16(
                false, a1, false, b1, (short)0, acc11, false, false);
        }

        const float wn0 = wnorm[t * 32 + l15];
        const float wn1 = wnorm[t * 32 + 16 + l15];
        const int n0 = t * 32 + l15;
        const int n1 = t * 32 + 16 + l15;
#pragma unroll
        for (int r = 0; r < 8; ++r) {
            float d00 = fmaf(-2.0f, acc00[r], wn0);
            float d01 = fmaf(-2.0f, acc01[r], wn1);
            float d10 = fmaf(-2.0f, acc10[r], wn0);
            float d11 = fmaf(-2.0f, acc11[r], wn1);
            if (d00 < best[0][r]) { best[0][r] = d00; bestn[0][r] = n0; }
            if (d01 < best[0][r]) { best[0][r] = d01; bestn[0][r] = n1; }
            if (d10 < best[1][r]) { best[1][r] = d10; bestn[1][r] = n0; }
            if (d11 < best[1][r]) { best[1][r] = d11; bestn[1][r] = n1; }
        }
        stage_b_wait(tid);
        __syncthreads();
    }

    // ---- cross-lane argmin within each 16-lane group ----
#pragma unroll
    for (int mt = 0; mt < 2; ++mt) {
#pragma unroll
        for (int r = 0; r < 8; ++r) {
            float bv = best[mt][r]; int bn = bestn[mt][r];
            for (int off = 8; off > 0; off >>= 1) {
                float ov = __shfl_xor(bv, off, 32);
                int   on = __shfl_xor(bn, off, 32);
                if (ov < bv || (ov == bv && on < bn)) { bv = ov; bn = on; }
            }
            if (l15 == 0) {
                int row = blockIdx.x * 128 + wave * 32 + mt * 16 + lh * 8 + r;
                idx_i[row] = bn;
                idx_f[row] = (float)bn;
            }
        }
    }
}

// =====================================================================
// Kernel 3: gather quantized rows, SSE loss partials, counts + segment sums
// =====================================================================
__global__ void vq_gather(const float* __restrict__ X,
                          const float* __restrict__ W,
                          const int* __restrict__ idx_i,
                          float* __restrict__ outQ,
                          float* __restrict__ sums,
                          float* __restrict__ counts,
                          float* __restrict__ accums) {
    __shared__ float red[256];
    const int row = blockIdx.x, tid = threadIdx.x;
    const int k = idx_i[row];
    float ls = 0.f;
    for (int d = tid; d < D_DIM; d += 256) {
        float q = W[(size_t)k * D_DIM + d];
        float x = X[(size_t)row * D_DIM + d];
        outQ[(size_t)row * D_DIM + d] = q;   // quantized_st forward value == q
        float df = q - x;
        ls += df * df;
        atomicAdd(&sums[(size_t)k * D_DIM + d], ONE_M_DECAY * x);
    }
    red[tid] = ls; __syncthreads();
    for (int s = 128; s > 0; s >>= 1) {
        if (tid < s) red[tid] += red[tid + s];
        __syncthreads();
    }
    if (tid == 0) {
        atomicAdd(&accums[0], red[0]);
        atomicAdd(&counts[k], 1.0f);
    }
}

// =====================================================================
// Kernel 4: reductions over K — sum(new_cs), entropy, usage
// =====================================================================
__global__ void vq_stats(const float* __restrict__ counts,
                         const float* __restrict__ ema_cs,
                         float* __restrict__ accums) {
    __shared__ float r0[256], r1[256], r2[256];
    const int tid = threadIdx.x;
    const int k = blockIdx.x * 256 + tid;
    float c = counts[k];
    float ncs = ema_cs[k] * DECAY + ONE_M_DECAY * c;
    float avg = c * (1.0f / (float)N_DIM);
    float ent = -avg * logf(avg + 1e-10f);
    float used = (c > 0.f) ? 1.f : 0.f;
    r0[tid] = ncs; r1[tid] = ent; r2[tid] = used;
    __syncthreads();
    for (int s = 128; s > 0; s >>= 1) {
        if (tid < s) { r0[tid] += r0[tid + s]; r1[tid] += r1[tid + s]; r2[tid] += r2[tid + s]; }
        __syncthreads();
    }
    if (tid == 0) {
        atomicAdd(&accums[1], r0[0]);
        atomicAdd(&accums[2], r1[0]);
        atomicAdd(&accums[3], r2[0]);
    }
}

// =====================================================================
// Kernel 5: EMA finalize + dead-code resampling (deterministic hash RNG)
// =====================================================================
__global__ void vq_final(const float* __restrict__ X,
                         const float* __restrict__ ema_cs,
                         const float* __restrict__ ema_w,
                         const float* __restrict__ counts,
                         const float* __restrict__ sums,
                         const float* __restrict__ accums,
                         float* __restrict__ outNW,
                         float* __restrict__ outNCS,
                         float* __restrict__ outNEW) {
    const int k = blockIdx.x, tid = threadIdx.x;
    const float n = accums[1];
    const float c = counts[k];
    const float ncs = ema_cs[k] * DECAY + ONE_M_DECAY * c;
    const bool dead = ncs < 2.0f;
    const float norm = (ncs + 1e-5f) / (n + (float)K_DIM * 1e-5f) * n;
    const float inv = 1.0f / norm;
    const unsigned rid = pcg(0x9E3779B9u ^ (unsigned)k) & (N_DIM - 1);
    for (int d = tid; d < D_DIM; d += 256) {
        float nev = ema_w[(size_t)k * D_DIM + d] * DECAY + sums[(size_t)k * D_DIM + d];
        float nw = nev * inv;
        if (dead) {
            unsigned h1 = pcg((unsigned)(k * D_DIM + d) * 2654435761u + 1u);
            unsigned h2 = pcg((unsigned)(k * D_DIM + d) * 2246822519u + 7u);
            float u1 = (float)(h1 >> 8) * (1.0f / 16777216.0f) + 1e-7f;
            float u2 = (float)(h2 >> 8) * (1.0f / 16777216.0f);
            float g = sqrtf(-2.0f * logf(u1)) * cosf(6.28318530718f * u2);
            float repl = X[(size_t)rid * D_DIM + d] + 0.01f * g;
            nev = repl; nw = repl;
        }
        outNW[(size_t)k * D_DIM + d] = nw;
        outNEW[(size_t)k * D_DIM + d] = nev;
    }
    if (tid == 0) outNCS[k] = dead ? 1.0f : ncs;
}

// =====================================================================
// Kernel 6: scalar outputs
// =====================================================================
__global__ void vq_scalars(const float* __restrict__ accums, float* __restrict__ out) {
    if (threadIdx.x == 0) {
        float mse = accums[0] * (1.0f / (float)((size_t)N_DIM * D_DIM));
        out[OFF_LOSS]   = (1.0f + BETA) * mse;  // codebook + BETA*commitment
        out[OFF_COMMIT] = mse;
        out[OFF_CODE]   = mse;
        out[OFF_PERP]   = expf(accums[2]);
        out[OFF_USAGE]  = accums[3] * (1.0f / (float)K_DIM);
    }
}

// =====================================================================
extern "C" void kernel_launch(void* const* d_in, const int* in_sizes, int n_in,
                              void* d_out, int out_size, void* d_ws, size_t ws_size,
                              hipStream_t stream) {
    (void)in_sizes; (void)n_in; (void)out_size; (void)ws_size;
    const float* X      = (const float*)d_in[0];
    const float* W      = (const float*)d_in[1];
    const float* ema_cs = (const float*)d_in[2];
    const float* ema_w  = (const float*)d_in[3];
    float* out = (float*)d_out;
    char*  ws  = (char*)d_ws;

    unsigned short* Wb   = (unsigned short*)(ws + WS_WB);
    float* wnorm  = (float*)(ws + WS_WN);
    float* sums   = (float*)(ws + WS_SUMS);
    float* counts = (float*)(ws + WS_CNT);
    float* accums = (float*)(ws + WS_ACC);
    int*   idx_i  = (int*)(ws + WS_IDX);

    // zero: sums + counts + accums (contiguous)
    (void)hipMemsetAsync(ws + WS_SUMS, 0,
                         (size_t)K_DIM * D_DIM * 4 + (size_t)K_DIM * 4 + 256, stream);

    const int smem_bytes = (128 * D_DIM + 2 * 32 * D_DIM) * 2;  // 294912
    (void)hipFuncSetAttribute((const void*)vq_argmin_kernel,
                              hipFuncAttributeMaxDynamicSharedMemorySize, smem_bytes);

    vq_prep_w<<<K_DIM, 256, 0, stream>>>(W, Wb, wnorm);
    vq_argmin_kernel<<<N_DIM / 128, 128, smem_bytes, stream>>>(
        X, Wb, wnorm, idx_i, out + OFF_IDX);
    vq_gather<<<N_DIM, 256, 0, stream>>>(X, W, idx_i, out + OFF_Q, sums, counts, accums);
    vq_stats<<<K_DIM / 256, 256, 0, stream>>>(counts, ema_cs, accums);
    vq_final<<<K_DIM, 256, 0, stream>>>(X, ema_cs, ema_w, counts, sums, accums,
                                        out + OFF_NW, out + OFF_NCS, out + OFF_NEW);
    vq_scalars<<<1, 32, 0, stream>>>(accums, out);
}
